// MultiHeadedAttentionGoogle_42382737277236
// MI455X (gfx1250) — matrix operand order repaired and checked by reading, not compiled, not run on hardware
//
#include <hip/hip_runtime.h>

// ---------------------------------------------------------------------------
// MI455X (gfx1250) multi-head attention, bf16 WMMA everywhere, f32 accumulate.
// B=4, S=2048, D=1024, H=16, dk=64.
// Software-pipelined: loads for step i+1 issue before WMMAs of step i.
// ---------------------------------------------------------------------------

#define DEVINL __device__ __forceinline__

typedef __attribute__((ext_vector_type(16))) __bf16 v16bf;
typedef __attribute__((ext_vector_type(8)))  float  v8f;

static constexpr int Bsz = 4;
static constexpr int S   = 2048;
static constexpr int D   = 1024;
static constexpr int H   = 16;
static constexpr int DK  = 64;
static constexpr int M   = Bsz * S;          // 8192 rows of activations

union BF16x16 {
  uint4 q[2];
  v16bf v;
};

// Build a v16bf A/B fragment from two 16-byte chunks (each 8 bf16 values).
DEVINL v16bf load2x8bf(const unsigned short* p0, const unsigned short* p1) {
  BF16x16 u;
  u.q[0] = *(const uint4*)p0;
  u.q[1] = *(const uint4*)p1;
  return u.v;
}

DEVINL v8f wmma_bf16(v16bf a, v16bf b, v8f c) {
  // (neg_a, A, neg_b, B, c_mod, C, reuse_a, reuse_b)
  return __builtin_amdgcn_wmma_f32_16x16x32_bf16(false, a, false, b,
                                                 (short)0, c, false, false);
}

DEVINL unsigned short f32_to_bf16_bits(float x) {
  return __builtin_bit_cast(unsigned short, (__bf16)x);
}

DEVINL unsigned int pack_bf16x2(float a, float b) {
  return (unsigned int)f32_to_bf16_bits(a) |
         ((unsigned int)f32_to_bf16_bits(b) << 16);
}

// ---------------------------------------------------------------------------
// fp32 -> bf16 convert (4 elements / thread)
// ---------------------------------------------------------------------------
__global__ void cvt_f32_to_bf16_x4(const float4* __restrict__ x,
                                   uint2* __restrict__ y, int n4) {
  int i = blockIdx.x * blockDim.x + threadIdx.x;
  if (i >= n4) return;
  float4 f = x[i];
  uint2 r;
  r.x = pack_bf16x2(f.x, f.y);
  r.y = pack_bf16x2(f.z, f.w);
  y[i] = r;
}

// ---------------------------------------------------------------------------
// GEMM: Y[M,1024] = Xbf[M,1024] @ Wbf^T + bias   (all matmuls in this problem)
//   workgroup: 256 threads = 8 waves; tile 128(M) x 64(N); K-loop step 32.
//   Each wave owns 16 M-rows and 4 WMMA accumulators across the 64 N cols.
//   Double-buffered A/B fragments so WMMAs overlap in-flight loads.
// MODE 0: store bf16, scaled by 1/sqrt(dk), layout [B,H,S,dk]   (Q)
// MODE 1: store bf16,                 layout [B,H,S,dk]         (K)
// MODE 2: store bf16,                 layout [B,H,dk,S]         (V^T)
// MODE 3: store f32 row-major [M,D]                             (out proj)
// ---------------------------------------------------------------------------
template <int MODE>
__global__ __launch_bounds__(256)
void gemm_bf16_wmma(const unsigned short* __restrict__ X,
                    const unsigned short* __restrict__ W,   // [N,K] row-major
                    const float* __restrict__ bias,
                    unsigned short* __restrict__ outbf,
                    float* __restrict__ outf) {
  const int lane = threadIdx.x & 31;
  const int wave = threadIdx.x >> 5;
  const int l15  = lane & 15;
  const int hi   = lane >> 4;                  // 0: lanes 0-15, 1: lanes 16-31

  const int m0 = blockIdx.y * 128 + wave * 16; // wave's 16 output rows
  const int n0 = blockIdx.x * 64;              // workgroup's 64 output cols

  v8f acc[4] = {};

  // A fragment: lane holds row (m0+l15); lanes>=16 take K+16..K+31.
  const unsigned short* xr = X + (size_t)(m0 + l15) * D + (hi ? 16 : 0);
  // B = W^T: column n of B is row (n0+nt*16+n) of W -> contiguous 32B.
  const unsigned short* wr[4];
#pragma unroll
  for (int nt = 0; nt < 4; ++nt)
    wr[nt] = W + (size_t)(n0 + nt * 16 + l15) * D + (hi ? 16 : 0);

  // prologue: fragments for k = 0
  v16bf a_cur = load2x8bf(xr, xr + 8);
  v16bf b_cur[4];
#pragma unroll
  for (int nt = 0; nt < 4; ++nt)
    b_cur[nt] = load2x8bf(wr[nt], wr[nt] + 8);

#pragma unroll 2
  for (int k = 32; k < D; k += 32) {
    // issue next step's 10 b128 loads first ...
    v16bf a_nxt = load2x8bf(xr + k, xr + k + 8);
    v16bf b_nxt[4];
#pragma unroll
    for (int nt = 0; nt < 4; ++nt)
      b_nxt[nt] = load2x8bf(wr[nt] + k, wr[nt] + k + 8);
    // ... then run current step's WMMAs while they are in flight
#pragma unroll
    for (int nt = 0; nt < 4; ++nt)
      acc[nt] = wmma_bf16(a_cur, b_cur[nt], acc[nt]);
    a_cur = a_nxt;
#pragma unroll
    for (int nt = 0; nt < 4; ++nt) b_cur[nt] = b_nxt[nt];
  }
#pragma unroll
  for (int nt = 0; nt < 4; ++nt)
    acc[nt] = wmma_bf16(a_cur, b_cur[nt], acc[nt]);

  // Epilogue. C layout: VGPR r holds (row r, col lane) / (row r+8, col lane-16)
  const int rbase = m0 + hi * 8;
#pragma unroll
  for (int nt = 0; nt < 4; ++nt) {
    const int col = n0 + nt * 16 + l15;
    const float bv = bias[col];
#pragma unroll
    for (int r = 0; r < 8; ++r) {
      const int mrow = rbase + r;
      float val = acc[nt][r] + bv;
      if (MODE == 3) {
        outf[(size_t)mrow * D + col] = val;
      } else {
        const int b = mrow >> 11;          // /2048
        const int s = mrow & (S - 1);
        const int h = col >> 6;            // /64
        const int d = col & (DK - 1);
        if (MODE == 0) val *= 0.125f;      // 1/sqrt(64) folded into Q
        size_t idx;
        if (MODE == 2)                     // V^T : [B,H,dk,S]
          idx = (((size_t)(b * H + h)) * DK + d) * S + s;
        else                               // Q,K : [B,H,S,dk]
          idx = (((size_t)(b * H + h)) * S + s) * DK + d;
        outbf[idx] = f32_to_bf16_bits(val);
      }
    }
  }
}

// ---------------------------------------------------------------------------
// Flash attention: one 16-query tile per wave, 8 waves/WG, 32-key steps.
// Q: [B,H,S,dk] bf16 (pre-scaled), K: [B,H,S,dk] bf16, Vt: [B,H,dk,S] bf16.
// O: [B,S,H*dk] bf16.
// K fragments are double-buffered across steps; V fragments issue before the
// softmax VALU block so global latency hides behind shuffle/exp work.
// ---------------------------------------------------------------------------
__global__ __launch_bounds__(256)
void attn_wmma(const unsigned short* __restrict__ Q,
               const unsigned short* __restrict__ K,
               const unsigned short* __restrict__ Vt,
               unsigned short* __restrict__ O) {
  __shared__ unsigned short p_lds[8][16][32];   // per-wave P staging (A-layout src)

  const int lane = threadIdx.x & 31;
  const int wave = threadIdx.x >> 5;
  const int l15  = lane & 15;
  const int hi   = lane >> 4;

  const int bh = blockIdx.y;                    // b*H + h
  const int q0 = blockIdx.x * 128 + wave * 16;

  const unsigned short* Qp = Q  + (size_t)bh * S * DK;
  const unsigned short* Kp = K  + (size_t)bh * S * DK;
  const unsigned short* Vp = Vt + (size_t)bh * DK * S;

  // Q A-fragments for d-chunks [0,32) and [32,64)
  const unsigned short* qr = Qp + (size_t)(q0 + l15) * DK + (hi ? 16 : 0);
  const v16bf qa0 = load2x8bf(qr,      qr + 8);
  const v16bf qa1 = load2x8bf(qr + 32, qr + 40);

  // K^T B-fragments for a 32-key step: [tile0 d0-31, tile0 d32-63, tile1 ...]
  auto loadK = [&](int kk, v16bf kb[4]) {
    const unsigned short* kr0 = Kp + (size_t)(kk + l15)      * DK + hi * 16;
    const unsigned short* kr1 = Kp + (size_t)(kk + 16 + l15) * DK + hi * 16;
    kb[0] = load2x8bf(kr0,      kr0 + 8);
    kb[1] = load2x8bf(kr0 + 32, kr0 + 40);
    kb[2] = load2x8bf(kr1,      kr1 + 8);
    kb[3] = load2x8bf(kr1 + 32, kr1 + 40);
  };

  float m[8], l[8];
#pragma unroll
  for (int r = 0; r < 8; ++r) { m[r] = -1e30f; l[r] = 0.0f; }
  v8f o[4] = {};

  v16bf kb[4];
  loadK(0, kb);

  for (int k0 = 0; k0 < S; k0 += 32) {
    // ---- scores: two 16x16 tiles (keys k0..+15 and k0+16..+31) ----
    v8f s0 = {}, s1 = {};
    s0 = wmma_bf16(qa0, kb[0], s0);
    s0 = wmma_bf16(qa1, kb[1], s0);
    s1 = wmma_bf16(qa0, kb[2], s1);
    s1 = wmma_bf16(qa1, kb[3], s1);

    // issue next step's K fragments and this step's V fragments NOW, so the
    // softmax VALU block below hides their latency.
    if (k0 + 32 < S) loadK(k0 + 32, kb);
    v16bf vb[4];
#pragma unroll
    for (int dt = 0; dt < 4; ++dt) {
      const unsigned short* vr =
          Vp + (size_t)(dt * 16 + l15) * S + k0 + hi * 16;
      vb[dt] = load2x8bf(vr, vr + 8);
    }
    if (k0 + 32 < S)                             // hint next V tile into cache
      __builtin_prefetch(Vp + (size_t)l15 * S + k0 + 32, 0, 0);

    // ---- online softmax; row stats live per-lane matching the C layout ----
    float p0[8], p1[8];
#pragma unroll
    for (int r = 0; r < 8; ++r) {
      float t = fmaxf(s0[r], s1[r]);
      t = fmaxf(t, __shfl_xor(t, 1));
      t = fmaxf(t, __shfl_xor(t, 2));
      t = fmaxf(t, __shfl_xor(t, 4));
      t = fmaxf(t, __shfl_xor(t, 8));
      const float mnew  = fmaxf(m[r], t);
      const float alpha = __expf(m[r] - mnew);
      const float e0 = __expf(s0[r] - mnew);
      const float e1 = __expf(s1[r] - mnew);
      float rs = e0 + e1;
      rs += __shfl_xor(rs, 1);
      rs += __shfl_xor(rs, 2);
      rs += __shfl_xor(rs, 4);
      rs += __shfl_xor(rs, 8);
      l[r] = l[r] * alpha + rs;
      m[r] = mnew;
      p0[r] = e0; p1[r] = e1;
      o[0][r] *= alpha; o[1][r] *= alpha; o[2][r] *= alpha; o[3][r] *= alpha;
    }

    // ---- restage P (C layout) -> LDS -> A layout ----
#pragma unroll
    for (int r = 0; r < 8; ++r) {
      p_lds[wave][r + hi * 8][l15]      = f32_to_bf16_bits(p0[r]);
      p_lds[wave][r + hi * 8][l15 + 16] = f32_to_bf16_bits(p1[r]);
    }
    asm volatile("s_wait_dscnt 0" ::: "memory");
    const unsigned short* pr = &p_lds[wave][l15][hi * 16];
    const v16bf pa = load2x8bf(pr, pr + 8);
    asm volatile("" ::: "memory");

    // ---- P @ V : four 16-wide d tiles (V fragments already in flight) ----
#pragma unroll
    for (int dt = 0; dt < 4; ++dt)
      o[dt] = wmma_bf16(pa, vb[dt], o[dt]);
  }

  // ---- normalize and store O as [B, S, H*dk] bf16 ----
  const int b = bh >> 4;
  const int h = bh & 15;
#pragma unroll
  for (int r = 0; r < 8; ++r) {
    const float rl = 1.0f / l[r];
    const int q = q0 + r + hi * 8;
    const size_t rowoff = ((size_t)b * S + q) * D + h * DK + l15;
#pragma unroll
    for (int dt = 0; dt < 4; ++dt)
      O[rowoff + dt * 16] = f32_to_bf16_bits(o[dt][r] * rl);
  }
}

// ---------------------------------------------------------------------------
extern "C" void kernel_launch(void* const* d_in, const int* in_sizes, int n_in,
                              void* d_out, int out_size, void* d_ws,
                              size_t ws_size, hipStream_t stream) {
  const float* q  = (const float*)d_in[0];
  const float* k  = (const float*)d_in[1];
  const float* v  = (const float*)d_in[2];
  const float* Wq = (const float*)d_in[3];
  const float* bq = (const float*)d_in[4];
  const float* Wk = (const float*)d_in[5];
  const float* bk = (const float*)d_in[6];
  const float* Wv = (const float*)d_in[7];
  const float* bv = (const float*)d_in[8];
  const float* Wo = (const float*)d_in[9];
  const float* bo = (const float*)d_in[10];
  float* out = (float*)d_out;

  // ---- workspace carve-up (bf16 elements) ----
  const size_t MX = (size_t)M * D;        // 8.39M elems per activation tensor
  const size_t MW = (size_t)D * D;
  unsigned short* ws  = (unsigned short*)d_ws;
  unsigned short* Xq  = ws; ws += MX;
  unsigned short* Xk  = ws; ws += MX;
  unsigned short* Xv  = ws; ws += MX;
  unsigned short* Wqb = ws; ws += MW;
  unsigned short* Wkb = ws; ws += MW;
  unsigned short* Wvb = ws; ws += MW;
  unsigned short* Wob = ws; ws += MW;
  unsigned short* Qb  = ws; ws += MX;
  unsigned short* Kb  = ws; ws += MX;
  unsigned short* Vtb = ws; ws += MX;
  unsigned short* Ob  = ws; ws += MX;

  // ---- 1) fp32 -> bf16 converts ----
  {
    const int nx4 = (int)(MX / 4), nw4 = (int)(MW / 4);
    dim3 blk(256);
    cvt_f32_to_bf16_x4<<<dim3((nx4 + 255) / 256), blk, 0, stream>>>(
        (const float4*)q, (uint2*)Xq, nx4);
    cvt_f32_to_bf16_x4<<<dim3((nx4 + 255) / 256), blk, 0, stream>>>(
        (const float4*)k, (uint2*)Xk, nx4);
    cvt_f32_to_bf16_x4<<<dim3((nx4 + 255) / 256), blk, 0, stream>>>(
        (const float4*)v, (uint2*)Xv, nx4);
    cvt_f32_to_bf16_x4<<<dim3((nw4 + 255) / 256), blk, 0, stream>>>(
        (const float4*)Wq, (uint2*)Wqb, nw4);
    cvt_f32_to_bf16_x4<<<dim3((nw4 + 255) / 256), blk, 0, stream>>>(
        (const float4*)Wk, (uint2*)Wkb, nw4);
    cvt_f32_to_bf16_x4<<<dim3((nw4 + 255) / 256), blk, 0, stream>>>(
        (const float4*)Wv, (uint2*)Wvb, nw4);
    cvt_f32_to_bf16_x4<<<dim3((nw4 + 255) / 256), blk, 0, stream>>>(
        (const float4*)Wo, (uint2*)Wob, nw4);
  }

  // ---- 2) QKV projections (WMMA GEMM, fused head-split + V transpose) ----
  {
    dim3 grid(D / 64, M / 128), blk(256);
    gemm_bf16_wmma<0><<<grid, blk, 0, stream>>>(Xq, Wqb, bq, Qb,  nullptr);
    gemm_bf16_wmma<1><<<grid, blk, 0, stream>>>(Xk, Wkb, bk, Kb,  nullptr);
    gemm_bf16_wmma<2><<<grid, blk, 0, stream>>>(Xv, Wvb, bv, Vtb, nullptr);
  }

  // ---- 3) flash attention ----
  {
    dim3 grid(S / 128, Bsz * H), blk(256);
    attn_wmma<<<grid, blk, 0, stream>>>(Qb, Kb, Vtb, Ob);
  }

  // ---- 4) output projection, fp32 result ----
  {
    dim3 grid(D / 64, M / 128), blk(256);
    gemm_bf16_wmma<3><<<grid, blk, 0, stream>>>(Ob, Wob, bo, nullptr, out);
  }
}
